// LLaDALlamaBlock_36120674959554
// MI455X (gfx1250) — matrix operand order repaired and checked by reading, not compiled, not run on hardware
//
#include <hip/hip_runtime.h>
#include <hip/hip_bf16.h>
#include <math.h>

// ---------------------------------------------------------------------------
// LLaDA/Llama block for MI455X (gfx1250, wave32, WMMA).
// - All GEMMs on v_wmma_f32_16x16x32_bf16, fp32 accumulate.
// - Activation intermediates in HBM as bf16; fp32 weights converted to bf16
//   while staging into LDS.
// - GEMM is double-buffered: async b128 copies (ASYNCcnt, saddr form) for the
//   bf16 A tile and register-staged weight loads for the B tile are issued
//   right after the per-iteration barrier and overlap with the WMMA stream.
// - Attention is flash-style (online softmax): logits never hit HBM.
// ---------------------------------------------------------------------------

typedef __attribute__((ext_vector_type(16))) __bf16 bf16x16;
typedef __attribute__((ext_vector_type(8)))  __bf16 bf16x8;
typedef __attribute__((ext_vector_type(4)))  __bf16 bf16x4;
typedef __attribute__((ext_vector_type(8)))  float  v8f;

#define WMMA_BF16(A, B, C) \
  __builtin_amdgcn_wmma_f32_16x16x32_bf16(false, (A), false, (B), (short)0, (C), false, false)

static __device__ __forceinline__ v8f v8f_zero() {
  v8f z = {0.f, 0.f, 0.f, 0.f, 0.f, 0.f, 0.f, 0.f};
  return z;
}

// Async copy 16B global -> LDS, GVS form: 32-bit per-lane offset + SGPR base.
// Generic pointers to LDS carry the byte offset in their low 32 bits.
static __device__ __forceinline__ void async_b128_s(unsigned ldsoff, unsigned voff,
                                                    unsigned long long sbase) {
  asm volatile("global_load_async_to_lds_b128 %0, %1, %2" ::"v"(ldsoff), "v"(voff), "s"(sbase)
               : "memory");
}
static __device__ __forceinline__ void wait_async0() {
  asm volatile("s_wait_asynccnt 0x0" ::: "memory");
}
static __device__ __forceinline__ unsigned lds_off(const void* p) {
  return (unsigned)(unsigned long long)p;
}

// ---------------------------------------------------------------------------
// RMSNorm: fp32 in -> bf16 out. One block (8 waves) per row of D=4096.
// ---------------------------------------------------------------------------
__global__ __launch_bounds__(256) void rmsnorm_kernel(const float* __restrict__ x,
                                                      const float* __restrict__ w,
                                                      __bf16* __restrict__ out, int D) {
  const int row = blockIdx.x;
  const float* xr = x + (size_t)row * D;
  __bf16* orow = out + (size_t)row * D;

  float ss = 0.f;
  for (int i = threadIdx.x; i < D; i += 256) {
    float v = xr[i];
    ss += v * v;
  }
  for (int m = 16; m >= 1; m >>= 1) ss += __shfl_xor(ss, m, 32);

  __shared__ float red[8];
  const int lane = threadIdx.x & 31, wv = threadIdx.x >> 5;
  if (lane == 0) red[wv] = ss;
  __syncthreads();
  if (wv == 0) {
    float v = (lane < 8) ? red[lane] : 0.f;
    for (int m = 4; m >= 1; m >>= 1) v += __shfl_xor(v, m, 32);
    if (lane == 0) red[0] = v;
  }
  __syncthreads();
  const float inv = rsqrtf(red[0] / (float)D + 1e-5f);
  for (int i = threadIdx.x; i < D; i += 256) orow[i] = (__bf16)(xr[i] * inv * w[i]);
}

// ---------------------------------------------------------------------------
// RoPE in place on bf16 q/k. Layout (B,T,H,HD), H=32, HD=128, half=64.
// ---------------------------------------------------------------------------
__global__ __launch_bounds__(256) void rope_kernel(__bf16* __restrict__ q, __bf16* __restrict__ k,
                                                   const float* __restrict__ s_,
                                                   const float* __restrict__ c_) {
  const unsigned idx = blockIdx.x * 256u + threadIdx.x;  // < B*T*H*64
  const unsigned d = idx & 63u;
  const unsigned h = (idx >> 6) & 31u;
  const unsigned bt = idx >> 11;
  const float s = s_[bt * 64u + d];
  const float c = c_[bt * 64u + d];
  const size_t base = ((size_t)bt * 32u + h) * 128u + d;

  float a1 = (float)q[base], a2 = (float)q[base + 64];
  q[base] = (__bf16)(a1 * c - a2 * s);
  q[base + 64] = (__bf16)(a2 * c + a1 * s);
  float b1 = (float)k[base], b2 = (float)k[base + 64];
  k[base] = (__bf16)(b1 * c - b2 * s);
  k[base + 64] = (__bf16)(b2 * c + b1 * s);
}

// ---------------------------------------------------------------------------
// GEMM:  C[M,N] = A[M,K](bf16) * B[K,N](fp32 weights)
// Block tile 128x128, k-step 32, double-buffered LDS (40 KB).
// Pipeline per k-step: store staged B regs -> LDS; wait_asynccnt; barrier;
// issue next A async copies + next B global loads; 8 WMMAs per wave.
// EPI: 0 plain, 1 C = aux(f32) + acc, 2 C = silu(aux(bf16)) * acc.
// ---------------------------------------------------------------------------
template <int EPI, typename CT>
__global__ __launch_bounds__(256) void gemm_kernel(const __bf16* __restrict__ A,
                                                   const float* __restrict__ Bw,
                                                   CT* __restrict__ C,
                                                   const void* __restrict__ aux_,
                                                   int M, int N, int K) {
  __shared__ __align__(16) __bf16 As[2][128][40];  // [buf][m][k], 80B rows
  __shared__ __align__(16) __bf16 Bs[2][128][40];  // [buf][n][k] transposed

  const int n0 = blockIdx.x * 128;
  const int m0 = blockIdx.y * 128;
  const int tid = threadIdx.x;
  const int wv = tid >> 5, lane = tid & 31;
  const int wr = wv >> 2, wc = wv & 3;  // 2x4 wave grid
  const int hi = lane >> 4, ln = lane & 15;

  // A tile async params: 2 x b128 per thread, rows ar0 and ar0+64, col ac0.
  const int ar0 = tid >> 2, ac0 = (tid & 3) << 3;
  const unsigned voffA0 = (unsigned)(ar0 * K + ac0) * 2u;
  const unsigned voffA1 = (unsigned)((ar0 + 64) * K + ac0) * 2u;
  const unsigned long long sA = (unsigned long long)(A + (size_t)m0 * K);
  unsigned ldsA0[2], ldsA1[2];
  ldsA0[0] = lds_off(&As[0][ar0][ac0]);
  ldsA0[1] = lds_off(&As[1][ar0][ac0]);
  ldsA1[0] = lds_off(&As[0][ar0 + 64][ac0]);
  ldsA1[1] = lds_off(&As[1][ar0 + 64][ac0]);

  // B tile coords: fixed n, k4 = bk + 8*i for i=0..3.
  const int bn = tid & 127;
  const int bk = ((tid >> 7) & 1) << 2;
  const float* const bbase = Bw + (size_t)bk * N + (n0 + bn);

  v8f acc[4][2];
  for (int tr = 0; tr < 4; ++tr)
    for (int tc = 0; tc < 2; ++tc) acc[tr][tc] = v8f_zero();

  const int KT = K >> 5;
  float rb[4][4];

  // --- prologue: tile 0 in flight ---
  async_b128_s(ldsA0[0], voffA0, sA);
  async_b128_s(ldsA1[0], voffA1, sA);
#pragma unroll
  for (int i = 0; i < 4; ++i)
#pragma unroll
    for (int j = 0; j < 4; ++j) rb[i][j] = bbase[(size_t)(8 * i + j) * N];

  int cur = 0;
  for (int kt = 0; kt < KT; ++kt, cur ^= 1) {
    // (1) store staged B regs (tile kt) to LDS, transposed [n][k].
#pragma unroll
    for (int i = 0; i < 4; ++i) {
      bf16x4 pk;
      pk.x = (__bf16)rb[i][0];
      pk.y = (__bf16)rb[i][1];
      pk.z = (__bf16)rb[i][2];
      pk.w = (__bf16)rb[i][3];
      *(bf16x4*)&Bs[cur][bn][bk + 8 * i] = pk;
    }
    // (2) tile kt fully resident after this.
    wait_async0();
    __syncthreads();
    // (3) issue next tile (kt+1) into the buffer freed by compute(kt-1).
    if (kt + 1 < KT) {
      const size_t koff = (size_t)(kt + 1) << 5;
      async_b128_s(ldsA0[cur ^ 1], voffA0, sA + koff * 2);
      async_b128_s(ldsA1[cur ^ 1], voffA1, sA + koff * 2);
      const float* bp = bbase + koff * N;
#pragma unroll
      for (int i = 0; i < 4; ++i)
#pragma unroll
        for (int j = 0; j < 4; ++j) rb[i][j] = bp[(size_t)(8 * i + j) * N];
      __builtin_prefetch(bp + (size_t)32 * N, 0, 0);  // tile kt+2 hint
    }
    // (4) 8 WMMAs on buffer `cur`.
#pragma unroll
    for (int tr = 0; tr < 4; ++tr) {
      bf16x16 a;
      const __bf16* ap = &As[cur][wr * 64 + tr * 16 + ln][0];
      ((bf16x8*)&a)[0] = *(const bf16x8*)(ap + 8 * hi);
      ((bf16x8*)&a)[1] = *(const bf16x8*)(ap + 16 + 8 * hi);
#pragma unroll
      for (int tc = 0; tc < 2; ++tc) {
        bf16x16 b;
        const __bf16* bp2 = &Bs[cur][wc * 32 + tc * 16 + ln][0];
        ((bf16x8*)&b)[0] = *(const bf16x8*)(bp2 + 8 * hi);
        ((bf16x8*)&b)[1] = *(const bf16x8*)(bp2 + 16 + 8 * hi);
        acc[tr][tc] = WMMA_BF16(a, b, acc[tr][tc]);
      }
    }
  }

  // Epilogue: lane holds col = ln, VGPR i -> row = i + 8*hi.
#pragma unroll
  for (int tr = 0; tr < 4; ++tr)
#pragma unroll
    for (int tc = 0; tc < 2; ++tc)
#pragma unroll
      for (int i = 0; i < 8; ++i) {
        const int row = m0 + wr * 64 + tr * 16 + 8 * hi + i;
        const int col = n0 + wc * 32 + tc * 16 + ln;
        const size_t o = (size_t)row * N + col;
        float v = acc[tr][tc][i];
        if (EPI == 1) v += ((const float*)aux_)[o];
        if (EPI == 2) {
          const float g = (float)((const __bf16*)aux_)[o];
          v *= g / (1.f + __expf(-g));  // silu(g) * acc
        }
        C[o] = (CT)v;
      }
}

// ---------------------------------------------------------------------------
// Flash attention (online softmax), all-bf16 I/O. One block per
// (b, h, 64-row q tile); s-tiles of 64. Q/K tiles via async b128 (saddr form);
// V transposed via packed b64 stores. ~59 KB LDS.
// ---------------------------------------------------------------------------
__global__ __launch_bounds__(256) void attn_kernel(const __bf16* __restrict__ q,
                                                   const __bf16* __restrict__ k,
                                                   const __bf16* __restrict__ v,
                                                   __bf16* __restrict__ out) {
  constexpr int T = 1024, H = 32, HD = 128;
  __shared__ __align__(16) __bf16 Qs[64][128];  // [m][d]
  __shared__ __align__(16) __bf16 Ks[64][128];  // [s][d]  (B for Q*K^T)
  __shared__ __align__(16) __bf16 Vt[128][64];  // [d][s]  (B for P*V)
  __shared__ __align__(16) __bf16 Pt[64][64];   // [m][s]  (A for P*V)
  __shared__ float redm[64][2];
  __shared__ float redl[64][2];
  __shared__ float rowm[64], rowl[64], rowfac[64];

  const int tid = threadIdx.x;
  const int lane = tid & 31, wv = tid >> 5;
  const int hi = lane >> 4, ln = lane & 15;
  const int qt = blockIdx.x & 15;
  const int h = (blockIdx.x >> 4) & 31;
  const int b = blockIdx.x >> 9;

  const int swr = wv >> 1, swc = wv & 1;  // S (64x64): 4x2 waves, 1x2 tiles
  const int owr = wv >> 2, owc = wv & 3;  // O (64x128): 2x4 waves, 2x2 tiles
  const float scale = 0.08838834764831845f;  // 1/sqrt(128)

  const __bf16* qbase = q + (((size_t)b * T + qt * 64) * H + h) * HD;
  const __bf16* kbase = k + ((size_t)b * T * H + h) * HD;
  const __bf16* vbase = v + ((size_t)b * T * H + h) * HD;
  constexpr size_t RS = (size_t)H * HD;  // element stride between t steps

  // Per-thread async copy params for 64x128 bf16 tiles: 4 x b128 per thread,
  // rows r0 + 16*i, fixed 16B column chunk.
  const int r0 = tid >> 4, qc = (tid & 15) << 3;
  const unsigned voffT = (unsigned)((size_t)r0 * RS + qc) * 2u;
  const unsigned rstep16 = (unsigned)(16 * RS * 2);  // 16 rows of stride RS

#pragma unroll
  for (int i = 0; i < 4; ++i)
    async_b128_s(lds_off(&Qs[r0 + 16 * i][qc]), voffT + i * rstep16,
                 (unsigned long long)qbase);
  if (tid < 64) {
    rowm[tid] = -3.0e38f;
    rowl[tid] = 0.f;
  }

  v8f O[2][2];
  for (int tr = 0; tr < 2; ++tr)
    for (int tc = 0; tc < 2; ++tc) O[tr][tc] = v8f_zero();

  for (int s0 = 0; s0 < T; s0 += 64) {
    __syncthreads();  // prior iteration done with Ks/Vt/Pt
    // K tile (row major) via async b128, saddr advances with s0.
    const unsigned long long sk = (unsigned long long)kbase + (size_t)s0 * RS * 2;
#pragma unroll
    for (int i = 0; i < 4; ++i)
      async_b128_s(lds_off(&Ks[r0 + 16 * i][qc]), voffT + i * rstep16, sk);
    // V tile transposed: 4 strided b16 loads -> one b64 store.
#pragma unroll
    for (int i = 0; i < 8; ++i) {
      const int idx = tid + i * 256;  // 2048 items = 16 sgroups x 128 d
      const int d = idx & 127, s4 = (idx >> 7) << 2;
      const __bf16* vp = vbase + (size_t)(s0 + s4) * RS + d;
      bf16x4 pk;
      pk.x = vp[0];
      pk.y = vp[RS];
      pk.z = vp[2 * RS];
      pk.w = vp[3 * RS];
      *(bf16x4*)&Vt[d][s4] = pk;
    }
    wait_async0();
    __syncthreads();

    // --- S = scale * Q K^T (64x64) ---
    v8f S[2] = {v8f_zero(), v8f_zero()};
#pragma unroll
    for (int k0 = 0; k0 < 128; k0 += 32) {
      bf16x16 a;
      const __bf16* ap = &Qs[swr * 16 + ln][k0];
      ((bf16x8*)&a)[0] = *(const bf16x8*)(ap + 8 * hi);
      ((bf16x8*)&a)[1] = *(const bf16x8*)(ap + 16 + 8 * hi);
#pragma unroll
      for (int tc = 0; tc < 2; ++tc) {
        bf16x16 bm;
        const __bf16* bp = &Ks[swc * 32 + tc * 16 + ln][k0];
        ((bf16x8*)&bm)[0] = *(const bf16x8*)(bp + 8 * hi);
        ((bf16x8*)&bm)[1] = *(const bf16x8*)(bp + 16 + 8 * hi);
        S[tc] = WMMA_BF16(a, bm, S[tc]);
      }
    }

    // --- per-row max of this s-tile ---
    float pm[8];
#pragma unroll
    for (int i = 0; i < 8; ++i) {
      S[0][i] *= scale;
      S[1][i] *= scale;
      pm[i] = fmaxf(S[0][i], S[1][i]);
    }
    for (int m = 8; m >= 1; m >>= 1)
#pragma unroll
      for (int i = 0; i < 8; ++i) pm[i] = fmaxf(pm[i], __shfl_xor(pm[i], m, 32));
    if (ln == 0)
#pragma unroll
      for (int i = 0; i < 8; ++i) redm[swr * 16 + 8 * hi + i][swc] = pm[i];
    __syncthreads();
    if (tid < 64) {
      const float mt = fmaxf(redm[tid][0], redm[tid][1]);
      const float mold = rowm[tid];
      const float mnew = fmaxf(mold, mt);
      rowfac[tid] = __expf(mold - mnew);  // first iter: exp(-inf)=0
      rowm[tid] = mnew;
    }
    __syncthreads();

    // --- P = exp(S - m) -> LDS (bf16); per-row partial sums ---
    float ps[8];
#pragma unroll
    for (int i = 0; i < 8; ++i) {
      const int row = swr * 16 + 8 * hi + i;
      const float m = rowm[row];
      const float p0 = __expf(S[0][i] - m);
      const float p1 = __expf(S[1][i] - m);
      Pt[row][swc * 32 + ln] = (__bf16)p0;
      Pt[row][swc * 32 + 16 + ln] = (__bf16)p1;
      ps[i] = p0 + p1;
    }
    for (int m = 8; m >= 1; m >>= 1)
#pragma unroll
      for (int i = 0; i < 8; ++i) ps[i] += __shfl_xor(ps[i], m, 32);
    if (ln == 0)
#pragma unroll
      for (int i = 0; i < 8; ++i) redl[swr * 16 + 8 * hi + i][swc] = ps[i];

    // --- rescale running O ---
#pragma unroll
    for (int tr = 0; tr < 2; ++tr)
#pragma unroll
      for (int tc = 0; tc < 2; ++tc)
#pragma unroll
        for (int i = 0; i < 8; ++i) O[tr][tc][i] *= rowfac[owr * 32 + tr * 16 + 8 * hi + i];
    __syncthreads();
    if (tid < 64) rowl[tid] = rowl[tid] * rowfac[tid] + redl[tid][0] + redl[tid][1];

    // --- O += P (64x64) * V (64x128) ---
#pragma unroll
    for (int k0 = 0; k0 < 64; k0 += 32) {
#pragma unroll
      for (int tr = 0; tr < 2; ++tr) {
        bf16x16 a;
        const __bf16* ap = &Pt[owr * 32 + tr * 16 + ln][k0];
        ((bf16x8*)&a)[0] = *(const bf16x8*)(ap + 8 * hi);
        ((bf16x8*)&a)[1] = *(const bf16x8*)(ap + 16 + 8 * hi);
#pragma unroll
        for (int tc = 0; tc < 2; ++tc) {
          bf16x16 bm;
          const __bf16* bp = &Vt[owc * 32 + tc * 16 + ln][k0];
          ((bf16x8*)&bm)[0] = *(const bf16x8*)(bp + 8 * hi);
          ((bf16x8*)&bm)[1] = *(const bf16x8*)(bp + 16 + 8 * hi);
          O[tr][tc] = WMMA_BF16(a, bm, O[tr][tc]);
        }
      }
    }
  }
  __syncthreads();

  // Normalize and write out[b][t][h*128 + d] (bf16).
#pragma unroll
  for (int tr = 0; tr < 2; ++tr)
#pragma unroll
    for (int tc = 0; tc < 2; ++tc)
#pragma unroll
      for (int i = 0; i < 8; ++i) {
        const int row = owr * 32 + tr * 16 + 8 * hi + i;
        const int col = owc * 32 + tc * 16 + ln;
        const float val = O[tr][tc][i] / rowl[row];
        out[((size_t)b * T + qt * 64 + row) * 4096 + h * HD + col] = (__bf16)val;
      }
}

// ---------------------------------------------------------------------------
// Launcher. Workspace (bytes, NDb = BT*D*2 = 16 MiB), 160 MiB total:
//   xn[NDb] | q[NDb] | k[NDb] | v[NDb] | attn[NDb] | x2(f32)[2*NDb] | h[3*NDb]
// (q/k/v region is reused as the FFN gate buffer h_ff: 3*NDb == BT*FF*2.)
// ---------------------------------------------------------------------------
extern "C" void kernel_launch(void* const* d_in, const int* in_sizes, int n_in,
                              void* d_out, int out_size, void* d_ws, size_t ws_size,
                              hipStream_t stream) {
  const float* x = (const float*)d_in[0];
  const float* sin_ = (const float*)d_in[1];
  const float* cos_ = (const float*)d_in[2];
  const float* anw = (const float*)d_in[3];
  const float* fnw = (const float*)d_in[4];
  const float* wq = (const float*)d_in[5];
  const float* wk = (const float*)d_in[6];
  const float* wv = (const float*)d_in[7];
  const float* wo = (const float*)d_in[8];
  const float* wff = (const float*)d_in[9];
  const float* wup = (const float*)d_in[10];
  const float* wout = (const float*)d_in[11];
  float* out = (float*)d_out;

  constexpr int BT = 2048, D = 4096, FF = 12288;
  const size_t NDb = (size_t)BT * D * 2;  // bytes of one bf16 activation buffer
  char* base = (char*)d_ws;
  __bf16* xn = (__bf16*)(base);
  __bf16* qb = (__bf16*)(base + NDb);
  __bf16* kb = (__bf16*)(base + 2 * NDb);
  __bf16* vb = (__bf16*)(base + 3 * NDb);
  __bf16* attnb = (__bf16*)(base + 4 * NDb);
  float* x2 = (float*)(base + 5 * NDb);    // fp32 residual stream
  __bf16* hh = (__bf16*)(base + 7 * NDb);  // BT*FF bf16
  __bf16* hff = qb;                        // reuse q/k/v: 3*NDb == BT*FF*2

  const dim3 blk(256);
  const dim3 gdd(D / 128, BT / 128);
  const dim3 gdf(FF / 128, BT / 128);

  rmsnorm_kernel<<<BT, blk, 0, stream>>>(x, anw, xn, D);
  gemm_kernel<0, __bf16><<<gdd, blk, 0, stream>>>(xn, wq, qb, nullptr, BT, D, D);
  gemm_kernel<0, __bf16><<<gdd, blk, 0, stream>>>(xn, wk, kb, nullptr, BT, D, D);
  gemm_kernel<0, __bf16><<<gdd, blk, 0, stream>>>(xn, wv, vb, nullptr, BT, D, D);
  rope_kernel<<<(2 * 1024 * 32 * 64) / 256, blk, 0, stream>>>(qb, kb, sin_, cos_);
  attn_kernel<<<2 * 32 * 16, blk, 0, stream>>>(qb, kb, vb, attnb);
  gemm_kernel<1, float><<<gdd, blk, 0, stream>>>(attnb, wo, x2, x, BT, D, D);
  rmsnorm_kernel<<<BT, blk, 0, stream>>>(x2, fnw, xn, D);
  gemm_kernel<0, __bf16><<<gdf, blk, 0, stream>>>(xn, wff, hff, nullptr, BT, FF, D);
  gemm_kernel<2, __bf16><<<gdf, blk, 0, stream>>>(xn, wup, hh, hff, BT, FF, D);
  gemm_kernel<1, float><<<gdd, blk, 0, stream>>>(hh, wout, out, x2, BT, D, FF);
}
